// Attention_90357521973502
// MI455X (gfx1250) — compile-verified, matrix operand order checked by reading, not run on hardware
//
#include <hip/hip_runtime.h>

// Problem constants (match reference)
#define BB   2
#define SS   2048
#define DD   4096
#define NH_  32
#define NKV_ 8
#define HD_  128
#define NREP_ 4
#define MM   (BB*SS)        // 4096 token rows
#define KVD  (NKV_*HD_)     // 1024

typedef __bf16 v16bf __attribute__((ext_vector_type(16)));
typedef float  v8f   __attribute__((ext_vector_type(8)));

union Frag16 { v16bf v; unsigned short h[16]; unsigned int u[8]; };

__device__ __forceinline__ unsigned short f2bf(float f) {
  union { float f; unsigned int u; } c; c.f = f;
  unsigned int r = c.u + 0x7FFFu + ((c.u >> 16) & 1u);   // round-to-nearest-even
  return (unsigned short)(r >> 16);
}
__device__ __forceinline__ float bf2f(unsigned short b) {
  union { unsigned int u; float f; } c; c.u = ((unsigned int)b) << 16;
  return c.f;
}
// K-offset of dword j inside a 16x32 bf16 A/B fragment (ISA 7.12.2):
// lanes 0-15: kbase 0, lanes 16-31: kbase 8; dwords 0-3 -> K+{0,2,4,6}, dwords 4-7 -> K+{16,18,20,22}
__device__ __forceinline__ int koff(int j, int hi) {
  return (hi ? 8 : 0) + ((j < 4) ? (2 * j) : (16 + 2 * (j - 4)));
}

// ---------------- elementwise prep kernels ----------------

__global__ void k_cast_bf16(const float* __restrict__ in, unsigned short* __restrict__ out, int n) {
  int i = blockIdx.x * blockDim.x + threadIdx.x;
  if (i < n) out[i] = f2bf(in[i]);
}

// w (K x N row-major f32) -> wT (N x K row-major bf16)
__global__ void k_tcast_bf16(const float* __restrict__ w, unsigned short* __restrict__ wt, int N, int K) {
  int i = blockIdx.x * blockDim.x + threadIdx.x;
  if (i >= N * K) return;
  int nn = i / K, kk = i - nn * K;
  wt[i] = f2bf(w[(size_t)kk * N + nn]);
}

// RoPE in place on Q bf16, layout (b,s,h,hd) flattened as MM x (NH*HD)
__global__ void k_rope_q(unsigned short* __restrict__ q) {
  int p = blockIdx.x * blockDim.x + threadIdx.x;
  const int PAIRS = MM * NH_ * (HD_ / 2);
  if (p >= PAIRS) return;
  int hd2 = p % (HD_ / 2);
  int ms  = p / ((HD_ / 2) * NH_);
  int s   = ms % SS;
  size_t base = (size_t)p * 2;           // same flattening: (ms, h, hd2) * 2
  float inv = __powf(10000.0f, -(float)hd2 / 64.0f);
  float ang = (float)s * inv;
  float c = cosf(ang), sn = sinf(ang);
  float x0 = bf2f(q[base]), x1 = bf2f(q[base + 1]);
  q[base]     = f2bf(x0 * c - x1 * sn);
  q[base + 1] = f2bf(x0 * sn + x1 * c);
}

// RoPE on K bf16 and relayout (b,s,g,hd) -> (b,g,s,hd)
__global__ void k_rope_k(const unsigned short* __restrict__ kin, unsigned short* __restrict__ kout) {
  int p = blockIdx.x * blockDim.x + threadIdx.x;
  const int PAIRS = MM * NKV_ * (HD_ / 2);
  if (p >= PAIRS) return;
  int hd2 = p % (HD_ / 2);
  int g   = (p / (HD_ / 2)) % NKV_;
  int ms  = p / ((HD_ / 2) * NKV_);
  int s   = ms % SS;
  int b   = ms / SS;
  size_t ib = (size_t)p * 2;
  float inv = __powf(10000.0f, -(float)hd2 / 64.0f);
  float ang = (float)s * inv;
  float c = cosf(ang), sn = sinf(ang);
  float x0 = bf2f(kin[ib]), x1 = bf2f(kin[ib + 1]);
  size_t ob = ((size_t)(b * NKV_ + g) * SS + s) * HD_ + 2 * hd2;
  kout[ob]     = f2bf(x0 * c - x1 * sn);
  kout[ob + 1] = f2bf(x0 * sn + x1 * c);
}

// V relayout (b,s,g,hd) -> (b,g,hd,s)   (transposed so P@V B-fragments are dword-contiguous)
__global__ void k_vtrans(const unsigned short* __restrict__ vin, unsigned short* __restrict__ vout) {
  int i = blockIdx.x * blockDim.x + threadIdx.x;
  if (i >= MM * KVD) return;
  int hd = i % HD_;
  int g  = (i / HD_) % NKV_;
  int ms = i / KVD;
  int s  = ms % SS;
  int b  = ms / SS;
  vout[((size_t)(b * NKV_ + g) * HD_ + hd) * SS + s] = vin[i];
}

// ---------------- bf16 WMMA GEMM: C(MxN) = A(MxK) @ Bt(NxK)^T ----------------
// One wave per block; each wave computes a 32x64 tile (2 M-subtiles x 4 N-subtiles,
// 8 WMMAs per K-step; every B fragment reused for 2 WMMAs, every A fragment for 4).
template <int OUT_BF16>
__global__ void __launch_bounds__(32, 1)
k_gemm_bf16_nt(const unsigned short* __restrict__ A,
               const unsigned short* __restrict__ Bt,
               unsigned short* __restrict__ Cb,
               float* __restrict__ Cf,
               int N, int K) {
  int lane = threadIdx.x & 31;
  int ln = lane & 15;
  int hi = lane >> 4;
  int m0 = blockIdx.y * 32;
  int n0 = blockIdx.x * 64;

  const unsigned short* arow0 = A + (size_t)(m0 + ln) * K;
  const unsigned short* arow1 = A + (size_t)(m0 + 16 + ln) * K;
  v8f acc[2][4] = {};

  for (int k0 = 0; k0 < K; k0 += 32) {
    if (k0 + 32 < K) {
      __builtin_prefetch((const void*)(arow0 + k0 + 32), 0, 0);
      __builtin_prefetch((const void*)(arow1 + k0 + 32), 0, 0);
    }
    Frag16 a0, a1;
#pragma unroll
    for (int j = 0; j < 8; ++j) {
      a0.u[j] = *(const unsigned int*)(arow0 + k0 + koff(j, hi));
      a1.u[j] = *(const unsigned int*)(arow1 + k0 + koff(j, hi));
    }
#pragma unroll
    for (int t = 0; t < 4; ++t) {
      const unsigned short* brow = Bt + (size_t)(n0 + t * 16 + ln) * K;
      Frag16 b;
#pragma unroll
      for (int j = 0; j < 8; ++j)
        b.u[j] = *(const unsigned int*)(brow + k0 + koff(j, hi));
      acc[0][t] = __builtin_amdgcn_wmma_f32_16x16x32_bf16(false, a0.v, false, b.v,
                                                          (short)0, acc[0][t], false, false);
      acc[1][t] = __builtin_amdgcn_wmma_f32_16x16x32_bf16(false, a1.v, false, b.v,
                                                          (short)0, acc[1][t], false, false);
    }
  }
#pragma unroll
  for (int m = 0; m < 2; ++m)
#pragma unroll
    for (int t = 0; t < 4; ++t)
#pragma unroll
      for (int i = 0; i < 8; ++i) {
        size_t r = (size_t)(m0 + m * 16 + i + hi * 8);
        size_t c = (size_t)(n0 + t * 16 + ln);
        if (OUT_BF16) Cb[r * N + c] = f2bf(acc[m][t][i]);
        else          Cf[r * N + c] = acc[m][t][i];
      }
}

// ---------------- flash attention (causal, GQA) ----------------
// grid: (SS/64, NH, B); block: 128 (4 waves). Wave w owns 16 query rows.
// Q: (b,s,h,hd) bf16;  K: (b,g,s,hd) bf16;  Vt: (b,g,hd,s) bf16;  O: (b,s,h,hd) bf16
// Q tile lives in LDS (not registers) to keep VGPR pressure under the spill threshold;
// persistent register state is just accv (64 VGPRs) + m_i/l_i (16).
__global__ void __launch_bounds__(128, 1)
k_attn(const unsigned short* __restrict__ Q,
       const unsigned short* __restrict__ Kc,
       const unsigned short* __restrict__ Vt,
       unsigned short* __restrict__ O) {
  __shared__ unsigned short lds_q[4 * 16 * HD_];  // per-wave 16x128 bf16 Q tile (16 KB)
  __shared__ unsigned short lds_p[4 * 16 * 32];   // per-wave 16x32 bf16 P tile (4 KB)

  int wave = threadIdx.x >> 5;
  int lane = threadIdx.x & 31;
  int ln = lane & 15;
  int hi = lane >> 4;
  int h = blockIdx.y;
  int b = blockIdx.z;
  int g = h / NREP_;
  int q0 = blockIdx.x * 64 + wave * 16;

  const float scale = 0.08838834764831845f;  // 1/sqrt(128)

  unsigned short* qlds = &lds_q[wave * 16 * HD_];
  unsigned short* pl   = &lds_p[wave * 16 * 32];

  // stage this wave's 16x128 Q tile into LDS: lane handles row ln, half-row hi*64
  {
    const unsigned short* qsrc = Q + (((size_t)b * SS + (q0 + ln)) * NH_ + h) * HD_ + hi * 64;
    unsigned short* qdst = qlds + ln * HD_ + hi * 64;
#pragma unroll
    for (int j = 0; j < 8; ++j)
      *(uint4*)(qdst + j * 8) = *(const uint4*)(qsrc + j * 8);
  }

  const unsigned short* kbase = Kc + (size_t)(b * NKV_ + g) * SS * HD_;
  const unsigned short* vbase = Vt + (size_t)(b * NKV_ + g) * HD_ * SS;

  float m_i[8], l_i[8];
  v8f accv[8] = {};
#pragma unroll
  for (int i = 0; i < 8; ++i) { m_i[i] = -1e30f; l_i[i] = 0.0f; }

  int kend = q0 + 16;                       // need keys <= q0+15 (causal)

  for (int kb = 0; kb < kend; kb += 32) {
    // ---- S = Q @ K^T for 32 keys (two 16x16 tiles); Q fragments re-read from LDS ----
    v8f sc0 = {}, sc1 = {};
#pragma unroll
    for (int c = 0; c < 4; ++c) {
      Frag16 aq;
#pragma unroll
      for (int j = 0; j < 8; ++j)
        aq.u[j] = *(const unsigned int*)(qlds + ln * HD_ + c * 32 + koff(j, hi));
      {
        const unsigned short* kr = kbase + (size_t)(kb + ln) * HD_;
        Frag16 bk;
#pragma unroll
        for (int j = 0; j < 8; ++j)
          bk.u[j] = *(const unsigned int*)(kr + c * 32 + koff(j, hi));
        sc0 = __builtin_amdgcn_wmma_f32_16x16x32_bf16(false, aq.v, false, bk.v,
                                                      (short)0, sc0, false, false);
      }
      {
        const unsigned short* kr = kbase + (size_t)(kb + 16 + ln) * HD_;
        Frag16 bk;
#pragma unroll
        for (int j = 0; j < 8; ++j)
          bk.u[j] = *(const unsigned int*)(kr + c * 32 + koff(j, hi));
        sc1 = __builtin_amdgcn_wmma_f32_16x16x32_bf16(false, aq.v, false, bk.v,
                                                      (short)0, sc1, false, false);
      }
    }

    // ---- fused online-softmax row loop: stats + accumulator rescale + P -> LDS ----
    int c0 = kb + ln, c1 = kb + 16 + ln;
#pragma unroll
    for (int i = 0; i < 8; ++i) {
      int r = q0 + i + hi * 8;
      float s0 = sc0[i] * scale; if (c0 > r) s0 = -1e30f;
      float s1 = sc1[i] * scale; if (c1 > r) s1 = -1e30f;
      float mx = fmaxf(s0, s1);
#pragma unroll
      for (int off = 1; off < 16; off <<= 1) mx = fmaxf(mx, __shfl_xor(mx, off, 16));
      float mn = fmaxf(m_i[i], mx);
      float p0 = __expf(s0 - mn);
      float p1 = __expf(s1 - mn);
      float rs = p0 + p1;
#pragma unroll
      for (int off = 1; off < 16; off <<= 1) rs += __shfl_xor(rs, off, 16);
      float alpha = __expf(m_i[i] - mn);
      l_i[i] = l_i[i] * alpha + rs;
      m_i[i] = mn;
#pragma unroll
      for (int f = 0; f < 8; ++f) accv[f][i] *= alpha;
      pl[(i + hi * 8) * 32 + ln]      = f2bf(p0);
      pl[(i + hi * 8) * 32 + 16 + ln] = f2bf(p1);
    }

    // ---- P (LDS, row-major 16x32) -> A-layout fragment ----
    Frag16 ap;
#pragma unroll
    for (int j = 0; j < 8; ++j)
      ap.u[j] = *(const unsigned int*)(pl + ln * 32 + koff(j, hi));

    // ---- O += P @ V  (V transposed: rows = hd, contiguous over s) ----
#pragma unroll
    for (int f = 0; f < 8; ++f) {
      const unsigned short* vr = vbase + (size_t)(f * 16 + ln) * SS + kb;
      Frag16 bv;
#pragma unroll
      for (int j = 0; j < 8; ++j)
        bv.u[j] = *(const unsigned int*)(vr + koff(j, hi));
      accv[f] = __builtin_amdgcn_wmma_f32_16x16x32_bf16(false, ap.v, false, bv.v,
                                                        (short)0, accv[f], false, false);
    }
  }

  // ---- normalize + store (b,s,h,hd) bf16 ----
#pragma unroll
  for (int f = 0; f < 8; ++f)
#pragma unroll
    for (int i = 0; i < 8; ++i) {
      int r = q0 + i + hi * 8;
      int col = f * 16 + ln;
      O[(((size_t)b * SS + r) * NH_ + h) * HD_ + col] = f2bf(accv[f][i] / l_i[i]);
    }
}

// ---------------- host-side orchestration ----------------
extern "C" void kernel_launch(void* const* d_in, const int* in_sizes, int n_in,
                              void* d_out, int out_size, void* d_ws, size_t ws_size,
                              hipStream_t stream) {
  const float* x  = (const float*)d_in[0];
  const float* wq = (const float*)d_in[1];
  const float* wk = (const float*)d_in[2];
  const float* wv = (const float*)d_in[3];
  const float* wo = (const float*)d_in[4];
  // d_in[5] = start_pos (0), d_in[6] = inference — reference builds RoPE tables from t=0.
  float* out = (float*)d_out;

  char* p = (char*)d_ws;
  unsigned short* xbf  = (unsigned short*)p; p += (size_t)MM * DD * 2;
  unsigned short* wqT  = (unsigned short*)p; p += (size_t)DD * DD * 2;
  unsigned short* wkT  = (unsigned short*)p; p += (size_t)KVD * DD * 2;
  unsigned short* wvT  = (unsigned short*)p; p += (size_t)KVD * DD * 2;
  unsigned short* woT  = (unsigned short*)p; p += (size_t)DD * DD * 2;
  unsigned short* qbf  = (unsigned short*)p; p += (size_t)MM * DD * 2;   // (b,s,h,hd), RoPE in place
  unsigned short* ktmp = (unsigned short*)p; p += (size_t)MM * KVD * 2;  // (b,s,g,hd)
  unsigned short* vtmp = (unsigned short*)p; p += (size_t)MM * KVD * 2;  // (b,s,g,hd)
  unsigned short* kbf  = (unsigned short*)p; p += (size_t)MM * KVD * 2;  // (b,g,s,hd)
  unsigned short* vtbf = (unsigned short*)p; p += (size_t)MM * KVD * 2;  // (b,g,hd,s)
  unsigned short* abf  = (unsigned short*)p; p += (size_t)MM * DD * 2;   // attention out (b,s,h,hd)

  // 1. cast activations + transpose-cast weights to bf16
  {
    int n = MM * DD;
    k_cast_bf16<<<(n + 255) / 256, 256, 0, stream>>>(x, xbf, n);
    k_tcast_bf16<<<(DD * DD + 255) / 256, 256, 0, stream>>>(wq, wqT, DD, DD);
    k_tcast_bf16<<<(KVD * DD + 255) / 256, 256, 0, stream>>>(wk, wkT, KVD, DD);
    k_tcast_bf16<<<(KVD * DD + 255) / 256, 256, 0, stream>>>(wv, wvT, KVD, DD);
    k_tcast_bf16<<<(DD * DD + 255) / 256, 256, 0, stream>>>(wo, woT, DD, DD);
  }

  // 2. QKV projections (WMMA bf16, f32 accumulate, bf16 out)
  k_gemm_bf16_nt<1><<<dim3(DD / 64, MM / 32), 32, 0, stream>>>(xbf, wqT, qbf, nullptr, DD, DD);
  k_gemm_bf16_nt<1><<<dim3(KVD / 64, MM / 32), 32, 0, stream>>>(xbf, wkT, ktmp, nullptr, KVD, DD);
  k_gemm_bf16_nt<1><<<dim3(KVD / 64, MM / 32), 32, 0, stream>>>(xbf, wvT, vtmp, nullptr, KVD, DD);

  // 3. RoPE + attention-friendly relayouts
  {
    int n = MM * NH_ * (HD_ / 2);
    k_rope_q<<<(n + 255) / 256, 256, 0, stream>>>(qbf);
    n = MM * NKV_ * (HD_ / 2);
    k_rope_k<<<(n + 255) / 256, 256, 0, stream>>>(ktmp, kbf);
    n = MM * KVD;
    k_vtrans<<<(n + 255) / 256, 256, 0, stream>>>(vtmp, vtbf);
  }

  // 4. causal flash attention
  k_attn<<<dim3(SS / 64, NH_, BB), 128, 0, stream>>>(qbf, kbf, vtbf, abf);

  // 5. output projection -> f32 d_out
  k_gemm_bf16_nt<0><<<dim3(DD / 64, MM / 32), 32, 0, stream>>>(abf, woT, nullptr, out, DD, DD);
}
